// GPT4ST_44092134260910
// MI455X (gfx1250) — matrix operand order repaired and checked by reading, not compiled, not run on hardware
//
#include <hip/hip_runtime.h>
#include <hip/hip_bf16.h>

typedef __attribute__((ext_vector_type(16))) _Float16 v16h;
typedef __attribute__((ext_vector_type(8)))  float    v8f;

constexpr int NB    = 64;    // batch
constexpr int NNODE = 170;   // N
constexpr int NT    = 12;    // T
constexpr int NC    = 192;   // C
constexpr int NH4   = 4;     // GAT heads
constexpr int NHC   = 48;    // per-head channels
constexpr int ND    = 768;   // GPT_D
constexpr int GHEADS= 12;
constexpr int GHD   = 64;
constexpr int NTIME = 288;
constexpr int NEDGE = NNODE*16;       // 2720
constexpr int NEALL = NEDGE + NNODE;  // 2890 (self loops appended)
constexpr int BN    = NB*NNODE;       // 10880 (multiple of 64)

// ---------------- async copy to LDS (CDNA5) with portable fallback ----------------
// Diagnostic-confirmed parameter type: (int __vector_size__(16) __device__ *) src,
// i.e. typed int4 pointers, global (AS1) src + LDS (AS3) dst, imm offset, imm cpol.
#if __has_builtin(__builtin_amdgcn_global_load_async_to_lds_b128)
typedef int i4v __attribute__((vector_size(16)));
typedef __attribute__((address_space(1))) i4v* gptr_i4;
typedef __attribute__((address_space(3))) i4v* lptr_i4;
#define CP16(lp, gp) __builtin_amdgcn_global_load_async_to_lds_b128( \
    (gptr_i4)(gp), (lptr_i4)(lp), 0, 0)
#if __has_builtin(__builtin_amdgcn_s_wait_asynccnt)
#define CP_WAIT() __builtin_amdgcn_s_wait_asynccnt(0)
#else
#define CP_WAIT() asm volatile("s_wait_asynccnt 0x0" ::: "memory")
#endif
#else
#define CP16(lp, gp) (*(float4*)(lp) = *(const float4*)(gp))
#define CP_WAIT()
#endif

// ---------------- helpers ----------------
__device__ inline float gelu_new(float x){
  float x3 = x*x*x;
  return 0.5f*x*(1.f + tanhf(0.7978845608028654f*(x + 0.044715f*x3)));
}
__device__ inline unsigned f2ord(float f){
  unsigned u = __float_as_uint(f);
  return (u & 0x80000000u) ? ~u : (u | 0x80000000u);
}
__device__ inline float ord2f(unsigned e){
  return (e & 0x80000000u) ? __uint_as_float(e ^ 0x80000000u) : __uint_as_float(~e);
}

// ---------------- generic WMMA GEMM ----------------
// C[M,N] = act(A[M,K] @ B[K,N] + bias) [+ resid] ; f32 in/out, f16 WMMA, f32 accum.
// 256 threads = 8 waves; block tile 64(M) x 64(N); BK=32; wave tile 16x32 (2 WMMAs).
template<int ACT, bool HAS_BIAS, bool HAS_RES>
__global__ __launch_bounds__(256) void k_gemm_wmma(
    const float* __restrict__ A, const float* __restrict__ Bw,
    const float* __restrict__ bias, const float* __restrict__ resid,
    float* __restrict__ Cout, int M, int N, int K)
{
  __shared__ __align__(16) float sA[64*32];   // [m][k] f32
  __shared__ __align__(16) float sB[32*64];   // [k][n] f32
  const int tid  = threadIdx.x;
  const int lane = tid & 31;
  const int wave = tid >> 5;       // 0..7
  const int wm   = wave & 3;       // 4 wave-tiles in M (16 rows each)
  const int wn   = wave >> 2;      // 2 wave-tiles in N (32 cols each)
  const int m0   = blockIdx.x * 64;
  const int n0   = blockIdx.y * 64;

  v8f acc0 = {0.f,0.f,0.f,0.f,0.f,0.f,0.f,0.f};
  v8f acc1 = {0.f,0.f,0.f,0.f,0.f,0.f,0.f,0.f};

  // ---- staging addresses: 2 x 16B chunks per thread per matrix ----
  const int ar0 = tid >> 3;              // rows 0..31
  const int ac0 = (tid & 7) * 4;         // col group
  const float* gA0 = A + (size_t)(m0 + ar0)*K + ac0;
  const float* gA1 = gA0 + (size_t)32*K; // rows 32..63
  float* lA0 = &sA[ar0*32 + ac0];
  float* lA1 = lA0 + 32*32;

  const int bk0 = tid >> 4;              // k rows 0..15
  const int bn4 = (tid & 15) * 4;
  const float* gB0 = Bw + (size_t)bk0*N + n0 + bn4;
  const float* gB1 = gB0 + (size_t)16*N; // k rows 16..31
  float* lB0 = &sB[bk0*64 + bn4];
  float* lB1 = lB0 + 16*64;

  // ---- fragment coordinates ----
  const int arow = wm*16 + (lane & 15);
  const int akb  = (lane < 16) ? 0 : 16;  // A K-base per lane group
  const int bcol = wn*32 + (lane & 15);
  const int bkb  = (lane < 16) ? 0 : 16;  // B K-base per lane group
  const size_t bstep = (size_t)32*N;

  for (int k0 = 0; k0 < K; k0 += 32) {
    CP16(lA0, gA0);
    CP16(lA1, gA1);
    CP16(lB0, gB0);
    CP16(lB1, gB1);
    if (k0 + 32 < K) {                    // prefetch next K-tile
      __builtin_prefetch(gA0 + 32, 0, 0);
      __builtin_prefetch(gA1 + 32, 0, 0);
      __builtin_prefetch(gB0 + bstep, 0, 0);
      __builtin_prefetch(gB1 + bstep, 0, 0);
    }
    CP_WAIT();
    __syncthreads();

    // A fragment: lane<16 -> K 0..15 ; lane>=16 -> K 16..31
    const float* pa = &sA[arow*32 + akb];
    v16h af, bf0, bf1;
    #pragma unroll
    for (int j = 0; j < 8; ++j) {
      af[j]   = (_Float16)pa[j];
      af[8+j] = (_Float16)pa[8+j];
    }
    // B fragments: 16 K-values for columns bcol and bcol+16
    #pragma unroll
    for (int m = 0; m < 16; ++m) {
      const float* pb = &sB[(bkb + m)*64 + bcol];
      bf0[m] = (_Float16)pb[0];
      bf1[m] = (_Float16)pb[16];
    }
    acc0 = __builtin_amdgcn_wmma_f32_16x16x32_f16(
        false, af, false, bf0, (short)0, acc0, false, false);
    acc1 = __builtin_amdgcn_wmma_f32_16x16x32_f16(
        false, af, false, bf1, (short)0, acc1, false, false);
    __syncthreads();

    gA0 += 32; gA1 += 32;
    gB0 += bstep; gB1 += bstep;
  }

  // ---- epilogue (branch-free: bias/resid presence templated) ----
  const int colg0 = n0 + wn*32 + (lane & 15);
  const int colg1 = colg0 + 16;
  const float bv0 = HAS_BIAS ? bias[colg0] : 0.f;
  const float bv1 = HAS_BIAS ? bias[colg1] : 0.f;
  #pragma unroll
  for (int r = 0; r < 8; ++r) {
    const int rowg = m0 + wm*16 + ((lane < 16) ? r : 8 + r);
    const size_t base = (size_t)rowg*N;
    float v0 = acc0[r] + bv0;
    float v1 = acc1[r] + bv1;
    if (ACT == 1) { v0 = fmaxf(v0, 0.f); v1 = fmaxf(v1, 0.f); }
    else if (ACT == 2) { v0 = gelu_new(v0); v1 = gelu_new(v1); }
    if (HAS_RES) { v0 += resid[base + colg0]; v1 += resid[base + colg1]; }
    Cout[base + colg0] = v0;
    Cout[base + colg1] = v1;
  }
}

// ---------------- embedding / start conv ----------------
__global__ __launch_bounds__(192) void k_embed(
    const float* __restrict__ hist, const float* __restrict__ sw,
    const float* __restrict__ sb, const float* __restrict__ tday,
    const float* __restrict__ tweek, float* __restrict__ token,
    float* __restrict__ tem)
{
  int bn = blockIdx.x;            // b*NNODE + n
  int b = bn / NNODE, n = bn % NNODE;
  int c = threadIdx.x;            // 0..191
  float acc = sb[c];
  #pragma unroll
  for (int f = 0; f < 3; ++f) {
    const float* hp = hist + ((size_t)(b*3 + f)*NNODE + n)*NT;
    const float* wp = sw + (size_t)c*(3*NT) + f*NT;
    #pragma unroll
    for (int t = 0; t < NT; ++t) acc += hp[t]*wp[t];
  }
  token[(size_t)bn*NC + c] = acc;
  float tod = hist[((size_t)(b*3 + 1)*NNODE + n)*NT + (NT-1)];
  int didx = (int)(tod * (float)NTIME);
  didx = max(0, min(NTIME-1, didx));
  int widx = (int)hist[((size_t)(b*3 + 2)*NNODE + n)*NT + (NT-1)];
  widx = max(0, min(6, widx));
  tem[(size_t)bn*NC + c] = tday[(size_t)didx*NC + c] + tweek[(size_t)widx*NC + c];
}

// ---------------- layer norm (one block per row) ----------------
__global__ __launch_bounds__(256) void k_layernorm(
    const float* __restrict__ x, const float* __restrict__ g,
    const float* __restrict__ bb, float* __restrict__ y, int C)
{
  __shared__ float red[256];
  int row = blockIdx.x, tid = threadIdx.x;
  const float* xr = x + (size_t)row*C;
  float s = 0.f;
  for (int c = tid; c < C; c += 256) s += xr[c];
  red[tid] = s; __syncthreads();
  for (int o = 128; o > 0; o >>= 1) { if (tid < o) red[tid] += red[tid+o]; __syncthreads(); }
  float mean = red[0] / C; __syncthreads();
  float v = 0.f;
  for (int c = tid; c < C; c += 256) { float d = xr[c]-mean; v += d*d; }
  red[tid] = v; __syncthreads();
  for (int o = 128; o > 0; o >>= 1) { if (tid < o) red[tid] += red[tid+o]; __syncthreads(); }
  float rstd = rsqrtf(red[0]/C + 1e-5f);
  float* yr = y + (size_t)row*C;
  for (int c = tid; c < C; c += 256) yr[c] = (xr[c]-mean)*rstd*g[c] + bb[c];
}

// ---------------- GAT ----------------
__global__ void k_gat_al(const float* __restrict__ H, const float* __restrict__ as,
                         const float* __restrict__ ad, float* __restrict__ als,
                         float* __restrict__ ald)
{
  int i = blockIdx.x*blockDim.x + threadIdx.x;
  if (i >= BN*NH4) return;
  int h = i & 3, bn = i >> 2;
  const float* hp = H + (size_t)bn*NC + h*NHC;
  const float* asp = as + h*NHC;
  const float* adp = ad + h*NHC;
  float s = 0.f, d = 0.f;
  #pragma unroll 8
  for (int c = 0; c < NHC; ++c) { float hv = hp[c]; s += hv*asp[c]; d += hv*adp[c]; }
  als[i] = s; ald[i] = d;
}

__global__ void k_fill_md(unsigned* __restrict__ mo, float* __restrict__ denom){
  int i = blockIdx.x*blockDim.x + threadIdx.x;
  if (i >= BN*NH4) return;
  mo[i] = f2ord(-1e30f);
  denom[i] = 0.f;
}

__global__ void k_fill_out(float* __restrict__ outb, const float* __restrict__ bias){
  int i = blockIdx.x*blockDim.x + threadIdx.x;
  if (i >= BN*NC) return;
  outb[i] = bias[i % NC];
}

__global__ void k_edge_logit(const int* __restrict__ src, const int* __restrict__ dst,
                             const float* __restrict__ als, const float* __restrict__ ald,
                             float* __restrict__ elog, unsigned* __restrict__ mo)
{
  int i = blockIdx.x*blockDim.x + threadIdx.x;
  if (i >= NB*NEALL*NH4) return;
  int h = i & 3; int e = (i >> 2) % NEALL; int b = (i >> 2) / NEALL;
  int s = (e < NEDGE) ? src[e] : (e - NEDGE);
  int d = (e < NEDGE) ? dst[e] : (e - NEDGE);
  float v = als[(size_t)(b*NNODE + s)*NH4 + h] + ald[(size_t)(b*NNODE + d)*NH4 + h];
  v = (v > 0.f) ? v : 0.2f*v;                 // leaky_relu 0.2
  elog[i] = v;
  atomicMax(&mo[(size_t)(b*NNODE + d)*NH4 + h], f2ord(v));
}

__global__ void k_edge_exp(const int* __restrict__ src, const int* __restrict__ dst,
                           const float* __restrict__ elog, const unsigned* __restrict__ mo,
                           float* __restrict__ denom, float* __restrict__ ex)
{
  int i = blockIdx.x*blockDim.x + threadIdx.x;
  if (i >= NB*NEALL*NH4) return;
  int h = i & 3; int e = (i >> 2) % NEALL; int b = (i >> 2) / NEALL;
  int d = (e < NEDGE) ? dst[e] : (e - NEDGE);
  float m = ord2f(mo[(size_t)(b*NNODE + d)*NH4 + h]);
  float xv = expf(elog[i] - m);
  ex[i] = xv;
  atomicAdd(&denom[(size_t)(b*NNODE + d)*NH4 + h], xv);
  (void)src;
}

__global__ void k_edge_scatter(const int* __restrict__ src, const int* __restrict__ dst,
                               const float* __restrict__ ex, const float* __restrict__ denom,
                               const float* __restrict__ H, float* __restrict__ outb)
{
  long long i = (long long)blockIdx.x*blockDim.x + threadIdx.x;
  if (i >= (long long)NB*NEALL*NC) return;
  int c = (int)(i % NC);
  int e = (int)((i / NC) % NEALL);
  int b = (int)(i / ((long long)NC*NEALL));
  int s = (e < NEDGE) ? src[e] : (e - NEDGE);
  int d = (e < NEDGE) ? dst[e] : (e - NEDGE);
  int h = c / NHC;
  float alpha = ex[(size_t)(b*NEALL + e)*NH4 + h] / denom[(size_t)(b*NNODE + d)*NH4 + h];
  atomicAdd(&outb[(size_t)(b*NNODE + d)*NC + c],
            alpha * H[(size_t)(b*NNODE + s)*NC + c]);
}

// ---------------- elementwise ----------------
__global__ void k_elu(const float* __restrict__ x, float* __restrict__ y, int n){
  int i = blockIdx.x*blockDim.x + threadIdx.x;
  if (i >= n) return;
  float v = x[i];
  y[i] = (v > 0.f) ? v : (expf(v) - 1.f);
}

__global__ void k_add(const float* __restrict__ a, const float* __restrict__ b,
                      float* __restrict__ y, int n){
  int i = blockIdx.x*blockDim.x + threadIdx.x;
  if (i >= n) return;
  y[i] = a[i] + b[i];
}

__global__ void k_concat(const float* __restrict__ tok, const float* __restrict__ nemb,
                         const float* __restrict__ tem, const float* __restrict__ ffn,
                         float* __restrict__ st)
{
  int i = blockIdx.x*blockDim.x + threadIdx.x;
  if (i >= BN*ND) return;
  int q = i % ND; int bn = i / ND; int n = bn % NNODE;
  float v;
  if (q < NC)          v = tok [(size_t)bn*NC + q];
  else if (q < 2*NC)   v = nemb[(size_t)n*NC + (q - NC)];
  else if (q < 3*NC)   v = tem [(size_t)bn*NC + (q - 2*NC)];
  else                 v = ffn [(size_t)bn*NC + (q - 3*NC)];
  st[i] = v;
}

__global__ void k_add_wpe(float* __restrict__ h, const float* __restrict__ wpe){
  int i = blockIdx.x*blockDim.x + threadIdx.x;
  if (i >= BN*ND) return;
  int q = i % ND; int bn = i / ND; int n = bn % NNODE;
  h[i] += wpe[(size_t)n*ND + q];
}

// ---------------- causal attention (scores in LDS) ----------------
__global__ __launch_bounds__(64) void k_attn(const float* __restrict__ qkv,
                                             float* __restrict__ ctx)
{
  int qi = blockIdx.x % NNODE;
  int bh = blockIdx.x / NNODE;
  int hh = bh % GHEADS; int b = bh / GHEADS;
  int tid = threadIdx.x;          // 64
  __shared__ float sc[NNODE];
  __shared__ float qv[GHD];
  __shared__ float red[64];
  const size_t rs = 3*ND;
  qv[tid] = qkv[((size_t)b*NNODE + qi)*rs + hh*GHD + tid];
  __syncthreads();
  float lmax = -1e30f;
  for (int k = tid; k <= qi; k += 64) {
    const float* kp = qkv + ((size_t)b*NNODE + k)*rs + ND + hh*GHD;
    float s = 0.f;
    #pragma unroll 8
    for (int d2 = 0; d2 < GHD; ++d2) s += qv[d2]*kp[d2];
    s *= 0.125f;                  // 1/sqrt(64)
    sc[k] = s;
    lmax = fmaxf(lmax, s);
  }
  red[tid] = lmax; __syncthreads();
  for (int o = 32; o > 0; o >>= 1) { if (tid < o) red[tid] = fmaxf(red[tid], red[tid+o]); __syncthreads(); }
  float mx = red[0]; __syncthreads();
  float lsum = 0.f;
  for (int k = tid; k <= qi; k += 64) { float e2 = expf(sc[k]-mx); sc[k] = e2; lsum += e2; }
  red[tid] = lsum; __syncthreads();
  for (int o = 32; o > 0; o >>= 1) { if (tid < o) red[tid] += red[tid+o]; __syncthreads(); }
  float inv = 1.f / red[0]; __syncthreads();
  float accv = 0.f;
  for (int k = 0; k <= qi; ++k) {
    accv += sc[k] * qkv[((size_t)b*NNODE + k)*rs + 2*ND + hh*GHD + tid];
  }
  ctx[((size_t)b*NNODE + qi)*ND + hh*GHD + tid] = accv * inv;
}

// ---------------- regression head + output transpose ----------------
__global__ void k_reg(const float* __restrict__ hf, const float* __restrict__ rw,
                      const float* __restrict__ rb, float* __restrict__ out)
{
  int i = blockIdx.x*blockDim.x + threadIdx.x;
  if (i >= BN*NT) return;
  int t = i % NT; int bn = i / NT;
  int b = bn / NNODE, n = bn % NNODE;
  const float* xr = hf + (size_t)bn*ND;
  float acc = rb[t];
  #pragma unroll 8
  for (int k = 0; k < ND; ++k) acc += xr[k]*rw[(size_t)k*NT + t];
  out[((size_t)b*NT + t)*NNODE + n] = acc;   // [B,12,N,1]
}

// ---------------- launch ----------------
#define GEMM(ACT_, HB_, HR_, A_, B_, BIAS_, RES_, C_, M_, N_, K_) \
  k_gemm_wmma<ACT_, HB_, HR_><<<dim3((M_)/64, (N_)/64), 256, 0, stream>>>(A_, B_, BIAS_, RES_, C_, M_, N_, K_)

static inline int cdiv(long long a, int b){ return (int)((a + b - 1)/b); }

extern "C" void kernel_launch(void* const* d_in, const int* in_sizes, int n_in,
                              void* d_out, int out_size, void* d_ws, size_t ws_size,
                              hipStream_t stream) {
  (void)in_sizes; (void)n_in; (void)out_size; (void)ws_size;
  const float* hist    = (const float*)d_in[0];
  const int*   eidx    = (const int*)  d_in[1];
  const float* start_w = (const float*)d_in[2];
  const float* start_b = (const float*)d_in[3];
  const float* time_day= (const float*)d_in[4];
  const float* time_wk = (const float*)d_in[5];
  const float* node_emb= (const float*)d_in[6];
  const float* ln1_g   = (const float*)d_in[7];
  const float* ln1_b   = (const float*)d_in[8];
  const float* ln2_g   = (const float*)d_in[9];
  const float* ln2_b   = (const float*)d_in[10];
  const float* ffn_w1  = (const float*)d_in[11];
  const float* ffn_b1  = (const float*)d_in[12];
  const float* ffn_w2  = (const float*)d_in[13];
  const float* ffn_b2  = (const float*)d_in[14];
  const float* gat1_w  = (const float*)d_in[15];
  const float* gat1_as = (const float*)d_in[16];
  const float* gat1_ad = (const float*)d_in[17];
  const float* gat1_b  = (const float*)d_in[18];
  const float* gat2_w  = (const float*)d_in[19];
  const float* gat2_as = (const float*)d_in[20];
  const float* gat2_ad = (const float*)d_in[21];
  const float* gat2_b  = (const float*)d_in[22];
  const float* fuse_w  = (const float*)d_in[23];
  const float* fuse_b  = (const float*)d_in[24];
  const float* wpe     = (const float*)d_in[25];
  const float* gln1_g  = (const float*)d_in[26];
  const float* gln1_b  = (const float*)d_in[27];
  const float* attn_w  = (const float*)d_in[28];
  const float* attn_b  = (const float*)d_in[29];
  const float* attn_pw = (const float*)d_in[30];
  const float* attn_pb = (const float*)d_in[31];
  const float* gln2_g  = (const float*)d_in[32];
  const float* gln2_b  = (const float*)d_in[33];
  const float* mlp_w1  = (const float*)d_in[34];
  const float* mlp_b1  = (const float*)d_in[35];
  const float* mlp_w2  = (const float*)d_in[36];
  const float* mlp_b2  = (const float*)d_in[37];
  const float* lnf_g   = (const float*)d_in[38];
  const float* lnf_b   = (const float*)d_in[39];
  const float* reg_w   = (const float*)d_in[40];
  const float* reg_b   = (const float*)d_in[41];

  const int* esrc = eidx;
  const int* edst = eidx + NEDGE;

  // ---- workspace carve-up (floats), heavy reuse ----
  float* ws = (float*)d_ws;
  size_t off = 0;
  auto alloc = [&](size_t n) { size_t o = off; off += (n + 63) & ~(size_t)63; return ws + o; };
  float* b192_0 = alloc((size_t)BN*NC);   // token
  float* b192_1 = alloc((size_t)BN*NC);   // tem
  float* b192_2 = alloc((size_t)BN*NC);   // ln_out -> elu(out1) -> gadd
  float* b192_3 = alloc((size_t)BN*NC);   // gat H -> skip2
  float* b192_4 = alloc((size_t)BN*NC);   // gat out -> ffn out
  float* b768_0 = alloc((size_t)BN*ND);   // ffn y1 -> fused+wpe (h) -> h3
  float* b768_1 = alloc((size_t)BN*ND);   // st -> ctx -> m_in -> hf
  float* b768_2 = alloc((size_t)BN*ND);   // a_in -> h2
  float* big    = alloc((size_t)BN*3072); // qkv (stride 2304) then mlp hidden (stride 3072)
  float* als    = alloc((size_t)BN*NH4);
  float* ald    = alloc((size_t)BN*NH4);
  unsigned* mo  = (unsigned*)alloc((size_t)BN*NH4);
  float* denom  = alloc((size_t)BN*NH4);
  float* elog   = alloc((size_t)NB*NEALL*NH4);
  float* ex     = alloc((size_t)NB*NEALL*NH4);

  const int TPB = 256;
  const int nE  = NB*NEALL*NH4;
  const long long nS = (long long)NB*NEALL*NC;

  // 1) start conv + temporal embedding
  k_embed<<<BN, 192, 0, stream>>>(hist, start_w, start_b, time_day, time_wk, b192_0, b192_1);
  // 2) ln1(token)
  k_layernorm<<<BN, 256, 0, stream>>>(b192_0, ln1_g, ln1_b, b192_2, NC);

  // 3) GAT layer 1 (input b192_2)
  GEMM(0, false, false, b192_2, gat1_w, (const float*)nullptr, (const float*)nullptr, b192_3, BN, NC, NC);
  k_gat_al<<<cdiv(BN*NH4, TPB), TPB, 0, stream>>>(b192_3, gat1_as, gat1_ad, als, ald);
  k_fill_md<<<cdiv(BN*NH4, TPB), TPB, 0, stream>>>(mo, denom);
  k_fill_out<<<cdiv(BN*NC, TPB), TPB, 0, stream>>>(b192_4, gat1_b);
  k_edge_logit<<<cdiv(nE, TPB), TPB, 0, stream>>>(esrc, edst, als, ald, elog, mo);
  k_edge_exp<<<cdiv(nE, TPB), TPB, 0, stream>>>(esrc, edst, elog, mo, denom, ex);
  k_edge_scatter<<<cdiv(nS, TPB), TPB, 0, stream>>>(esrc, edst, ex, denom, b192_3, b192_4);
  // elu -> layer2 input
  k_elu<<<cdiv(BN*NC, TPB), TPB, 0, stream>>>(b192_4, b192_2, BN*NC);

  // 4) GAT layer 2
  GEMM(0, false, false, b192_2, gat2_w, (const float*)nullptr, (const float*)nullptr, b192_3, BN, NC, NC);
  k_gat_al<<<cdiv(BN*NH4, TPB), TPB, 0, stream>>>(b192_3, gat2_as, gat2_ad, als, ald);
  k_fill_md<<<cdiv(BN*NH4, TPB), TPB, 0, stream>>>(mo, denom);
  k_fill_out<<<cdiv(BN*NC, TPB), TPB, 0, stream>>>(b192_4, gat2_b);
  k_edge_logit<<<cdiv(nE, TPB), TPB, 0, stream>>>(esrc, edst, als, ald, elog, mo);
  k_edge_exp<<<cdiv(nE, TPB), TPB, 0, stream>>>(esrc, edst, elog, mo, denom, ex);
  k_edge_scatter<<<cdiv(nS, TPB), TPB, 0, stream>>>(esrc, edst, ex, denom, b192_3, b192_4);

  // 5) gadd = gat_out + token ; skip2 = ln2(gadd)
  k_add<<<cdiv(BN*NC, TPB), TPB, 0, stream>>>(b192_4, b192_0, b192_2, BN*NC);
  k_layernorm<<<BN, 256, 0, stream>>>(b192_2, ln2_g, ln2_b, b192_3, NC);

  // 6) FFN: relu(skip2@w1+b1) @ w2 + b2 + gadd
  GEMM(1, true, false, b192_3, ffn_w1, ffn_b1, (const float*)nullptr, b768_0, BN, ND, NC);
  GEMM(0, true, true,  b768_0, ffn_w2, ffn_b2, b192_2, b192_4, BN, NC, ND);

  // 7) st = [token | node_emb | tem | ffn] ; fused = st@fuse_w+fuse_b ; h = fused + wpe
  k_concat<<<cdiv(BN*ND, TPB), TPB, 0, stream>>>(b192_0, node_emb, b192_1, b192_4, b768_1);
  GEMM(0, true, false, b768_1, fuse_w, fuse_b, (const float*)nullptr, b768_0, BN, ND, ND);
  k_add_wpe<<<cdiv(BN*ND, TPB), TPB, 0, stream>>>(b768_0, wpe);

  // 8) GPT block
  k_layernorm<<<BN, 256, 0, stream>>>(b768_0, gln1_g, gln1_b, b768_2, ND);
  GEMM(0, true, false, b768_2, attn_w, attn_b, (const float*)nullptr, big, BN, 3*ND, ND);
  k_attn<<<NB*GHEADS*NNODE, 64, 0, stream>>>(big, b768_1);
  GEMM(0, true, true, b768_1, attn_pw, attn_pb, b768_0, b768_2, BN, ND, ND);   // h2
  k_layernorm<<<BN, 256, 0, stream>>>(b768_2, gln2_g, gln2_b, b768_1, ND);
  GEMM(2, true, false, b768_1, mlp_w1, mlp_b1, (const float*)nullptr, big, BN, 4*ND, ND);
  GEMM(0, true, true, big, mlp_w2, mlp_b2, b768_2, b768_0, BN, ND, 4*ND);      // h3
  k_layernorm<<<BN, 256, 0, stream>>>(b768_0, lnf_g, lnf_b, b768_1, ND);

  // 9) regression head -> [B,12,N,1]
  k_reg<<<cdiv(BN*NT, TPB), TPB, 0, stream>>>(b768_1, reg_w, reg_b, (float*)d_out);
}
